// GraphTransformerLayer_30502857736724
// MI455X (gfx1250) — matrix operand order repaired and checked by reading, not compiled, not run on hardware
//
#include <hip/hip_runtime.h>
#include <hip/hip_bf16.h>

#define D   128   // d_model
#define GH  4     // graph heads
#define HC  128   // per-head channels
#define QKV 512   // GH*HC
#define FFN 512   // ffn hidden

typedef __attribute__((ext_vector_type(16))) __bf16 v16bf;
typedef __attribute__((ext_vector_type(8)))  __bf16 v8bf;
typedef __attribute__((ext_vector_type(8)))  float  v8f;

// ---------------------------------------------------------------------------
// f32 -> bf16 streaming convert (n multiple of 8)
// ---------------------------------------------------------------------------
__global__ void f32_to_bf16(const float* __restrict__ src, __bf16* __restrict__ dst,
                            long n)
{
  long i = ((long)blockIdx.x * blockDim.x + threadIdx.x) * 8;
  if (i >= n) return;
  float4 f0 = *(const float4*)(src + i);
  float4 f1 = *(const float4*)(src + i + 4);
  v8bf o;
  o[0] = (__bf16)f0.x; o[1] = (__bf16)f0.y; o[2] = (__bf16)f0.z; o[3] = (__bf16)f0.w;
  o[4] = (__bf16)f1.x; o[5] = (__bf16)f1.y; o[6] = (__bf16)f1.z; o[7] = (__bf16)f1.w;
  *(v8bf*)(dst + i) = o;
}

__device__ __forceinline__ v16bf load_b_frag(const __bf16* p) {
  v8bf b0 = *(const v8bf*)(p);
  v8bf b1 = *(const v8bf*)(p + 8);
  return __builtin_shufflevector(b0, b1, 0,1,2,3,4,5,6,7,8,9,10,11,12,13,14,15);
}

// ---------------------------------------------------------------------------
// Tiled WMMA GEMM: out[N,M] = act(A[N,K](bf16) @ Wb[M,K](bf16)^T + bias)
// Block = 128 threads = 4 waves -> 64x64 output tile:
//   wave w -> rows (blockIdx.x*4+w)*16..+15, 4 column sub-tiles of 16.
// W tile (64 x K bf16; 16KB for K=128, 64KB for K=512) is DMA'd into LDS with
// global_load_async_to_lds_b128 (ASYNCcnt path, no VGPR round-trip).
// A-fragment lane map (16-bit A 16x32, contiguous k order used here, not the ISA table order): m = lane&15, half = lane>>4,
//   elems 0..15 -> K = half*16 + e
//   => two contiguous v8bf (16B) loads at +0 and +8 bf16 elements.
// B-fragment lane map (dense 16-bit B 32x16): n = lane&15,
//   elems 0..15 -> K = (lane>>4)*16 + e.  B[k][n] = W row n of tile.
// D layout: col = lane&15, row = (lane>>4)*8 + r for acc element r.
// Outputs: C (f32, optional) and/or Cb (bf16, optional).
// ---------------------------------------------------------------------------
template <int K>
__global__ __launch_bounds__(128) void gemm_wmma_bf16(
    const __bf16* __restrict__ A, const __bf16* __restrict__ W,
    const float* __restrict__ bias, float* __restrict__ C,
    __bf16* __restrict__ Cb, int Nrows, int M, int relu)
{
  __shared__ __bf16 sW[64 * K];
  const int tid = threadIdx.x;
  const int cb0 = blockIdx.y * 64;          // first output column of block

  // Async DMA stage: 64 rows x K bf16 of W -> LDS (16B per lane per issue).
  {
    const __bf16* wsrc = W + (size_t)cb0 * K;
    const int total = 64 * K;               // bf16 elements, multiple of 1024
    for (int i = tid * 8; i < total; i += 128 * 8) {
      unsigned lds_off = (unsigned)(size_t)&sW[i];   // low 32b of flat = LDS byte addr
      asm volatile("global_load_async_to_lds_b128 %0, %1, off"
                   :: "v"(lds_off), "v"(wsrc + i) : "memory");
    }
    asm volatile("s_wait_asynccnt 0x0" ::: "memory");
  }
  __syncthreads();

  const int wave = tid >> 5;
  const int lane = tid & 31;
  const int rt   = blockIdx.x * 4 + wave;
  if (rt * 16 >= Nrows) return;            // wave-uniform guard (EXEC stays full)

  const int half = lane >> 4;
  const int l16  = lane & 15;

  v8f acc0 = {}, acc1 = {}, acc2 = {}, acc3 = {};
  const __bf16* arow = A + (size_t)(rt * 16 + l16) * K + half * 16;
  const int kSteps = K >> 5;
#pragma unroll 4
  for (int ks = 0; ks < kSteps; ++ks) {
    const __bf16* ap = arow + ks * 32;
    if (ks + 1 < kSteps) __builtin_prefetch(ap + 32, 0, 3);
    v8bf a0 = *(const v8bf*)(ap);
    v8bf a1 = *(const v8bf*)(ap + 8);
    v16bf af = __builtin_shufflevector(a0, a1, 0,1,2,3,4,5,6,7,8,9,10,11,12,13,14,15);

    const __bf16* bbase = &sW[(size_t)l16 * K + ks * 32 + half * 16];
    v16bf bf0 = load_b_frag(bbase);
    v16bf bf1 = load_b_frag(bbase + 16 * K);
    v16bf bf2 = load_b_frag(bbase + 32 * K);
    v16bf bf3 = load_b_frag(bbase + 48 * K);

    acc0 = __builtin_amdgcn_wmma_f32_16x16x32_bf16(false, af, false, bf0, (short)0, acc0, false, false);
    acc1 = __builtin_amdgcn_wmma_f32_16x16x32_bf16(false, af, false, bf1, (short)0, acc1, false, false);
    acc2 = __builtin_amdgcn_wmma_f32_16x16x32_bf16(false, af, false, bf2, (short)0, acc2, false, false);
    acc3 = __builtin_amdgcn_wmma_f32_16x16x32_bf16(false, af, false, bf3, (short)0, acc3, false, false);
  }

  const int row0 = rt * 16 + half * 8;
  v8f accs[4] = {acc0, acc1, acc2, acc3};
#pragma unroll
  for (int t = 0; t < 4; ++t) {
    const int col = cb0 + t * 16 + l16;
    const float bb = bias[col];
#pragma unroll
    for (int r = 0; r < 8; ++r) {
      float v = accs[t][r] + bb;
      if (relu) v = fmaxf(v, 0.f);
      if (C)  C[(size_t)(row0 + r) * M + col]  = v;
      if (Cb) Cb[(size_t)(row0 + r) * M + col] = (__bf16)v;
    }
  }
}

// ---------------------------------------------------------------------------
// Segment-softmax aggregation over edges
// ---------------------------------------------------------------------------
__global__ void init_seg(float* m, float* den, float* agg, int N) {
  int i = blockIdx.x * blockDim.x + threadIdx.x;
  if (i < N * GH) { m[i] = -3.402823466e38f; den[i] = 0.f; }
  if (i < N * QKV) agg[i] = 0.f;
}

__device__ __forceinline__ void atomic_max_f32(float* addr, float val) {
  if (val >= 0.f) atomicMax((int*)addr, __float_as_int(val));
  else            atomicMin((unsigned int*)addr, __float_as_uint(val));
}

__global__ void edge_logits(const float* __restrict__ q, const float* __restrict__ k,
                            const int* __restrict__ src, const int* __restrict__ dst,
                            float* __restrict__ alpha, float* __restrict__ m, int E)
{
  int idx = blockIdx.x * blockDim.x + threadIdx.x;
  if (idx >= E * GH) return;
  int e = idx >> 2, h = idx & 3;
  int s = src[e], d = dst[e];
  const float4* qp = (const float4*)(q + (size_t)d * QKV + h * HC);
  const float4* kp = (const float4*)(k + (size_t)s * QKV + h * HC);
  float acc = 0.f;
#pragma unroll 8
  for (int c = 0; c < HC / 4; ++c) {
    float4 a = qp[c], b = kp[c];
    acc += a.x * b.x + a.y * b.y + a.z * b.z + a.w * b.w;
  }
  acc *= 0.08838834764831845f;            // 1/sqrt(HC)
  alpha[idx] = acc;
  atomic_max_f32(&m[d * GH + h], acc);
}

__global__ void edge_exp(const int* __restrict__ dst, float* __restrict__ alpha,
                         const float* __restrict__ m, float* __restrict__ den, int E)
{
  int idx = blockIdx.x * blockDim.x + threadIdx.x;
  if (idx >= E * GH) return;
  int e = idx >> 2, h = idx & 3;
  int d = dst[e];
  float ex = expf(alpha[idx] - m[d * GH + h]);
  alpha[idx] = ex;                         // alpha buffer reused as ex
  atomicAdd(&den[d * GH + h], ex);
}

__global__ void edge_agg(const int* __restrict__ src, const int* __restrict__ dst,
                         const float* __restrict__ ex, const float* __restrict__ den,
                         const float* __restrict__ v, float* __restrict__ agg, int E)
{
  int idx = blockIdx.x * blockDim.x + threadIdx.x;
  if (idx >= E * GH) return;
  int e = idx >> 2, h = idx & 3;
  int s = src[e], d = dst[e];
  float w = ex[idx] / den[d * GH + h];
  const float4* vp = (const float4*)(v + (size_t)s * QKV + h * HC);
  float* ap = agg + (size_t)d * QKV + h * HC;
#pragma unroll 4
  for (int c = 0; c < HC / 4; ++c) {
    float4 vv = vp[c];
    atomicAdd(ap + c * 4 + 0, w * vv.x);
    atomicAdd(ap + c * 4 + 1, w * vv.y);
    atomicAdd(ap + c * 4 + 2, w * vv.z);
    atomicAdd(ap + c * 4 + 3, w * vv.w);
  }
}

// ---------------------------------------------------------------------------
// Fused residual + LayerNorm (one block = one node, 128 threads = 4 wave32)
// ---------------------------------------------------------------------------
__device__ __forceinline__ void reduce2_128(float& s, float& s2) {
#pragma unroll
  for (int o = 16; o > 0; o >>= 1) {
    s  += __shfl_down(s,  o, 32);
    s2 += __shfl_down(s2, o, 32);
  }
  __shared__ float r1[4], r2[4];
  int w = threadIdx.x >> 5;
  if ((threadIdx.x & 31) == 0) { r1[w] = s; r2[w] = s2; }
  __syncthreads();
  s  = r1[0] + r1[1] + r1[2] + r1[3];
  s2 = r2[0] + r2[1] + r2[2] + r2[3];
}

__global__ __launch_bounds__(128) void fuse_ln1(
    const float* __restrict__ x, const float* __restrict__ agg,
    const float* __restrict__ skip, const float* __restrict__ g,
    const float* __restrict__ b, float* __restrict__ out)
{
  int n = blockIdx.x, c = threadIdx.x;
  const float* ag = agg + (size_t)n * QKV;
  float gmean = 0.25f * (ag[c] + ag[c + 128] + ag[c + 256] + ag[c + 384]);
  float y = x[(size_t)n * D + c] + gmean + skip[(size_t)n * D + c];
  float s = y, s2 = y * y;
  reduce2_128(s, s2);
  float mu  = s * (1.f / D);
  float var = fmaxf(s2 * (1.f / D) - mu * mu, 0.f);
  out[(size_t)n * D + c] = (y - mu) * rsqrtf(var + 1e-5f) * g[c] + b[c];
}

__global__ __launch_bounds__(128) void add_ln(
    const float* __restrict__ a, const float* __restrict__ r,
    const float* __restrict__ g, const float* __restrict__ b,
    float* __restrict__ out, __bf16* __restrict__ outb)
{
  int n = blockIdx.x, c = threadIdx.x;
  float y = a[(size_t)n * D + c] + r[(size_t)n * D + c];
  float s = y, s2 = y * y;
  reduce2_128(s, s2);
  float mu  = s * (1.f / D);
  float var = fmaxf(s2 * (1.f / D) - mu * mu, 0.f);
  float o = (y - mu) * rsqrtf(var + 1e-5f) * g[c] + b[c];
  if (out)  out[(size_t)n * D + c]  = o;
  if (outb) outb[(size_t)n * D + c] = (__bf16)o;
}

// ---------------------------------------------------------------------------
extern "C" void kernel_launch(void* const* d_in, const int* in_sizes, int n_in,
                              void* d_out, int out_size, void* d_ws, size_t ws_size,
                              hipStream_t stream)
{
  const float* x     = (const float*)d_in[0];
  const float* te    = (const float*)d_in[1];
  const float* gq_w  = (const float*)d_in[2];  const float* gq_b = (const float*)d_in[3];
  const float* gk_w  = (const float*)d_in[4];  const float* gk_b = (const float*)d_in[5];
  const float* gv_w  = (const float*)d_in[6];  const float* gv_b = (const float*)d_in[7];
  const float* gs_w  = (const float*)d_in[8];  const float* gs_b = (const float*)d_in[9];
  // mha_wq/bq (10,11) and mha_wk/bk (12,13) are mathematically dead:
  // softmax over a single kv position is identically 1.
  const float* mha_wv = (const float*)d_in[14]; const float* mha_bv = (const float*)d_in[15];
  const float* mha_wo = (const float*)d_in[16]; const float* mha_bo = (const float*)d_in[17];
  const float* ffn_w1 = (const float*)d_in[18]; const float* ffn_b1 = (const float*)d_in[19];
  const float* ffn_w2 = (const float*)d_in[20]; const float* ffn_b2 = (const float*)d_in[21];
  const float* ln1_g  = (const float*)d_in[22]; const float* ln1_b  = (const float*)d_in[23];
  const float* ln2_g  = (const float*)d_in[24]; const float* ln2_b  = (const float*)d_in[25];
  const float* ln3_g  = (const float*)d_in[26]; const float* ln3_b  = (const float*)d_in[27];
  const int*   ei     = (const int*)d_in[28];

  const int N = in_sizes[0] / D;
  const int E = in_sizes[28] / 2;

  // Workspace carve-out
  float* ws    = (float*)d_ws;
  float* q     = ws;                        // N*512 f32
  float* kbuf  = q     + (size_t)N * QKV;   // N*512 f32
  float* vbuf  = kbuf  + (size_t)N * QKV;   // N*512 f32
  float* skip  = vbuf  + (size_t)N * QKV;   // N*128 f32
  float* alpha = skip  + (size_t)N * D;     // E*4  f32 (reused as ex)
  float* mbuf  = alpha + (size_t)E * GH;    // N*4  f32
  float* den   = mbuf  + (size_t)N * GH;    // N*4  f32
  float* agg   = den   + (size_t)N * GH;    // N*512 f32
  float* x1    = agg   + (size_t)N * QKV;   // N*128 f32
  float* t2    = x1    + (size_t)N * D;     // N*128 f32
  float* x2    = t2    + (size_t)N * D;     // N*128 f32
  float* f2    = x2    + (size_t)N * D;     // N*128 f32
  __bf16* xb   = (__bf16*)(f2 + (size_t)N * D); // N*128 bf16
  __bf16* teb  = xb  + (size_t)N * D;           // N*128 bf16
  __bf16* tvb  = teb + (size_t)N * D;           // N*128 bf16
  __bf16* x2b  = tvb + (size_t)N * D;           // N*128 bf16
  __bf16* hb   = x2b + (size_t)N * D;           // N*512 bf16
  // bf16 weights (converted once per launch)
  __bf16* gqb  = hb  + (size_t)N * FFN;         // 512*128
  __bf16* gkb  = gqb + QKV * D;
  __bf16* gvb  = gkb + QKV * D;
  __bf16* gsb  = gvb + QKV * D;                 // 128*128
  __bf16* wvb  = gsb + D * D;
  __bf16* wob  = wvb + D * D;
  __bf16* f1b  = wob + D * D;                   // 512*128
  __bf16* f2b  = f1b + FFN * D;                 // 128*512

  const int nRT      = (N + 15) / 16;
  const int rtBlocks = (nRT + 3) / 4;
  const dim3 blk(128);
  const long cvtN = (long)N * D;
  const int cvtBlocks = (int)((cvtN / 8 + 255) / 256);
  const int cvtBig = (QKV * D / 8 + 255) / 256;   // 512x128 weights
  const int cvtSml = (D * D / 8 + 255) / 256;     // 128x128 weights

  // --- weights -> bf16 (tiny, once per launch) ---
  f32_to_bf16<<<cvtBig, 256, 0, stream>>>(gq_w,   gqb, (long)QKV * D);
  f32_to_bf16<<<cvtBig, 256, 0, stream>>>(gk_w,   gkb, (long)QKV * D);
  f32_to_bf16<<<cvtBig, 256, 0, stream>>>(gv_w,   gvb, (long)QKV * D);
  f32_to_bf16<<<cvtSml, 256, 0, stream>>>(gs_w,   gsb, (long)D * D);
  f32_to_bf16<<<cvtSml, 256, 0, stream>>>(mha_wv, wvb, (long)D * D);
  f32_to_bf16<<<cvtSml, 256, 0, stream>>>(mha_wo, wob, (long)D * D);
  f32_to_bf16<<<cvtBig, 256, 0, stream>>>(ffn_w1, f1b, (long)FFN * D);
  f32_to_bf16<<<cvtBig, 256, 0, stream>>>(ffn_w2, f2b, (long)D * FFN);

  // --- activations -> bf16 (once) ---
  f32_to_bf16<<<cvtBlocks, 256, 0, stream>>>(x,  xb,  cvtN);
  f32_to_bf16<<<cvtBlocks, 256, 0, stream>>>(te, teb, cvtN);

  // --- graph q/k/v + root skip projections (WMMA) ---
  gemm_wmma_bf16<D><<<dim3(rtBlocks, QKV / 64), blk, 0, stream>>>(xb, gqb, gq_b, q,    nullptr, N, QKV, 0);
  gemm_wmma_bf16<D><<<dim3(rtBlocks, QKV / 64), blk, 0, stream>>>(xb, gkb, gk_b, kbuf, nullptr, N, QKV, 0);
  gemm_wmma_bf16<D><<<dim3(rtBlocks, QKV / 64), blk, 0, stream>>>(xb, gvb, gv_b, vbuf, nullptr, N, QKV, 0);
  gemm_wmma_bf16<D><<<dim3(rtBlocks, D   / 64), blk, 0, stream>>>(xb, gsb, gs_b, skip, nullptr, N, D,   0);

  // --- segment softmax over incoming edges ---
  init_seg<<<(N * QKV + 255) / 256, 256, 0, stream>>>(mbuf, den, agg, N);
  const int eThreads = E * GH;
  edge_logits<<<(eThreads + 255) / 256, 256, 0, stream>>>(q, kbuf, ei, ei + E, alpha, mbuf, E);
  edge_exp   <<<(eThreads + 255) / 256, 256, 0, stream>>>(ei + E, alpha, mbuf, den, E);
  edge_agg   <<<(eThreads + 255) / 256, 256, 0, stream>>>(ei, ei + E, alpha, den, vbuf, agg, E);

  // --- x1 = LN(x + mean_heads(agg) + skip) ---
  fuse_ln1<<<N, blk, 0, stream>>>(x, agg, skip, ln1_g, ln1_b, x1);

  // --- temporal branch (softmax==1): temp_out = (te@Wv^T+bv)@Wo^T+bo ---
  gemm_wmma_bf16<D><<<dim3(rtBlocks, D / 64), blk, 0, stream>>>(teb, wvb, mha_bv, nullptr, tvb, N, D, 0);
  gemm_wmma_bf16<D><<<dim3(rtBlocks, D / 64), blk, 0, stream>>>(tvb, wob, mha_bo, t2, nullptr, N, D, 0);
  add_ln<<<N, blk, 0, stream>>>(x1, t2, ln2_g, ln2_b, x2, x2b);

  // --- FFN ---
  gemm_wmma_bf16<D>  <<<dim3(rtBlocks, FFN / 64), blk, 0, stream>>>(x2b, f1b, ffn_b1, nullptr, hb, N, FFN, 1);
  gemm_wmma_bf16<FFN><<<dim3(rtBlocks, D   / 64), blk, 0, stream>>>(hb,  f2b, ffn_b2, f2, nullptr, N, D,  0);
  add_ln<<<N, blk, 0, stream>>>(x2, f2, ln3_g, ln3_b, (float*)d_out, nullptr);
}
